// RNNLangModel_34574486733070
// MI455X (gfx1250) — matrix operand order repaired and checked by reading, not compiled, not run on hardware
//
#include <hip/hip_runtime.h>

#define VOCAB  50257
#define EMBED  128
#define HIDDEN 512
#define BATCH  64
#define SEQ    512
#define KX     (HIDDEN + EMBED)   // 640 = concat(h, emb) width
#define KIT    (KX / 32)          // 20 K-iterations of 32
#define XSTR   648                // xbuf row stride (halfs): 1296B, 16B aligned, bank-skewed (+4 dw)

#define LDSK   9                  // K-iters 0..8  resident in LDS (303 KB)
#define REGK   10                 // K-iters 9..18 register-resident (80 VGPRs)
                                  // K-iter 19 streamed via WGP$ each step (32KB, L1-resident)
#define WSTR   296                // w1lds row stride in halfs: 592B, 16B aligned, bank-skewed (+20 dw)

typedef _Float16 h16;
typedef __attribute__((ext_vector_type(16))) _Float16     v16h;
typedef __attribute__((ext_vector_type(8)))  float        v8f;
typedef __attribute__((ext_vector_type(4)))  unsigned int u32x4;

union Frag {            // one 16x32 (A) or 32x16 (B) f16 WMMA fragment: 32B/lane
    u32x4 q[2];
    h16   h[16];
    v16h  v;
};
union FW {              // 16 fp32 loaded as 4x b128
    u32x4 q[4];
    float f[16];
};

// ---------------------------------------------------------------- prepass ---
__global__ void k_w1_to_f16(const float* __restrict__ w, h16* __restrict__ o, int n) {
    int i = blockIdx.x * blockDim.x + threadIdx.x;
    if (i < n) o[i] = (h16)w[i];
}

// --------------------------------------------------------------- RNN scan ---
// 4 blocks x 1024 threads; block b owns batch rows [16b,16b+16).
// Wave w (0..31) owns hidden columns [16w,16w+16).
// W1^T residency: 9 K-slices in LDS, 10 in VGPRs, 1 streamed through WGP$.
__global__ __launch_bounds__(1024)
void k_rnn(const int* __restrict__ tok,      // [B,S] int32
           const float* __restrict__ emb,    // [V,E]
           const h16* __restrict__ w1h,      // [H,KX] f16
           const float* __restrict__ b1,     // [H]
           h16* __restrict__ hlast)          // [B,H] f16 out
{
    __shared__ __align__(16) h16 xbuf[16 * XSTR];          //  20,736 B
    __shared__ __align__(16) h16 w1lds[HIDDEN * WSTR];     // 303,104 B  (total 323,840 <= 320KB LDS)

    const int tid  = threadIdx.x;
    const int lane = tid & 31;
    const int wave = tid >> 5;            // 0..31 -> N tile
    const int rb   = blockIdx.x * 16;     // batch row base

    // h_0 = 0
    for (int i = tid; i < 16 * HIDDEN; i += 1024)
        xbuf[(i >> 9) * XSTR + (i & 511)] = (h16)0.0f;

    // Fill LDS-resident W1 slices: w1lds[col*WSTR + k'], k' in [0, LDSK*32)
    for (int idx = tid; idx < HIDDEN * (LDSK * 32 / 8); idx += 1024) {   // 512*36 8-half chunks
        int col = idx / 36;
        int ck  = idx - col * 36;
        *(u32x4*)(w1lds + col * WSTR + ck * 8) =
            *(const u32x4*)(w1h + (size_t)col * KX + ck * 8);
    }

    // Register-resident B fragments for K-iters 9..18.
    // B[k,n] = W1[n,k]; lane = column, 16 contiguous K halfs (lanes 16-31: +16).
    const int   ncol = wave * 16 + (lane & 15);
    const int   bko  = (lane < 16) ? 0 : 16;
    const float bias = b1[ncol];
    Frag bfr[REGK];
#pragma unroll
    for (int r = 0; r < REGK; ++r) {
        const u32x4* p = (const u32x4*)(w1h + (size_t)ncol * KX + (LDSK + r) * 32 + bko);
        bfr[r].q[0] = p[0];
        bfr[r].q[1] = p[1];
    }
    // Streamed slice (K-iter 19): volatile so it is re-read each step (L1-hot),
    // not hoisted into registers (which would spill).
    const volatile u32x4* bsp =
        (const volatile u32x4*)(w1h + (size_t)ncol * KX + (LDSK + REGK) * 32 + bko);

    const int  m     = lane & 15;                  // A row
    const int  ako   = (lane < 16) ? 0 : 16;       // A K-chunk skew
    const int  rbase = (lane < 16) ? 0 : 8;        // C row base
    const h16* arow  = xbuf + m * XSTR;
    const h16* brow  = w1lds + ncol * WSTR;

    for (int t = 0; t < SEQ; ++t) {
        // kick off the streamed B slice early
        Frag bs;
        bs.q[0] = bsp[0];
        bs.q[1] = bsp[1];

        // stage embeddings for step t into x[:, 512:640]
#pragma unroll
        for (int i = tid; i < 16 * EMBED; i += 1024) {
            int r = i >> 7, cc = i & 127;
            int id = tok[(rb + r) * SEQ + t];
            xbuf[r * XSTR + HIDDEN + cc] = (h16)emb[(size_t)id * EMBED + cc];
        }
        __syncthreads();                           // x = [h_t | e_t] ready

        v8f c;
#pragma unroll
        for (int j = 0; j < 8; ++j) c[j] = 0.0f;

        // K-iters 0..8: B from LDS
#pragma unroll
        for (int kk = 0; kk < LDSK; ++kk) {
            Frag a, bl;
            a.q[0]  = *(const u32x4*)(arow + kk * 32 + ako);
            a.q[1]  = *(const u32x4*)(arow + kk * 32 + ako + 8);
            bl.q[0] = *(const u32x4*)(brow + kk * 32 + bko);
            bl.q[1] = *(const u32x4*)(brow + kk * 32 + bko + 8);
            c = __builtin_amdgcn_wmma_f32_16x16x32_f16(false, a.v, false, bl.v,
                                                       (short)0, c, false, false);
        }
        // K-iters 9..18: B from registers
#pragma unroll
        for (int r = 0; r < REGK; ++r) {
            const int kk = LDSK + r;
            Frag a;
            a.q[0] = *(const u32x4*)(arow + kk * 32 + ako);
            a.q[1] = *(const u32x4*)(arow + kk * 32 + ako + 8);
            c = __builtin_amdgcn_wmma_f32_16x16x32_f16(false, a.v, false, bfr[r].v,
                                                       (short)0, c, false, false);
        }
        // K-iter 19: streamed B
        {
            const int kk = LDSK + REGK;
            Frag a;
            a.q[0] = *(const u32x4*)(arow + kk * 32 + ako);
            a.q[1] = *(const u32x4*)(arow + kk * 32 + ako + 8);
            c = __builtin_amdgcn_wmma_f32_16x16x32_f16(false, a.v, false, bs.v,
                                                       (short)0, c, false, false);
        }
        __syncthreads();                           // all waves done reading x

        // h_{t+1}[rbase+r, ncol] = sigmoid(c[r] + b1)
#pragma unroll
        for (int r = 0; r < 8; ++r) {
            float s = 1.0f / (1.0f + __expf(-(c[r] + bias)));
            xbuf[(rbase + r) * XSTR + ncol] = (h16)s;
        }
        // h writes (cols 0..511) and next step's emb staging (cols 512..639)
        // are disjoint; the barrier at loop top orders both before compute.
    }

    __syncthreads();
    for (int i = tid; i < 16 * HIDDEN; i += 1024) {
        int r = i >> 9, cc = i & 511;
        hlast[(size_t)(rb + r) * HIDDEN + cc] = xbuf[r * XSTR + cc];
    }
}

// ----------------------------------------------------------------- logits ---
// HBM-bound stream of W2 (103 MB -> ~5us at 23.3 TB/s). Wave owns one 16-col
// vocab tile, computes all 4 M-tiles (B fragment reused 4x; h_last cache-hot).
// W2 fp32 -> f16 converted on the fly (avoids a 52MB prepass + extra traffic).
#define NT2 ((VOCAB + 15) / 16)   // 3142 vocab tiles

__global__ __launch_bounds__(256)
void k_logits(const h16* __restrict__ hlast,   // [B,H] f16
              const float* __restrict__ W2,    // [V,H]
              const float* __restrict__ b2,    // [V]
              float* __restrict__ out)         // [B,V]
{
    const int lane = threadIdx.x & 31;
    const int wave = threadIdx.x >> 5;
    const int tile = blockIdx.x * 8 + wave;
    if (tile >= NT2) return;                   // uniform per wave

    const int  v    = tile * 16 + (lane & 15); // this lane's vocab column
    const bool vok  = (v < VOCAB);
    const int  bko  = (lane < 16) ? 0 : 16;
    const int  m0   = lane & 15;
    const int  ako  = (lane < 16) ? 0 : 16;

    v8f c[4];
#pragma unroll
    for (int i = 0; i < 4; ++i)
#pragma unroll
        for (int j = 0; j < 8; ++j) c[i][j] = 0.0f;

    const float* wrow = W2 + (size_t)v * HIDDEN;

#pragma unroll 2
    for (int kk = 0; kk < HIDDEN / 32; ++kk) {
        FW bw;
        if (vok) {
            const u32x4* pb = (const u32x4*)(wrow + kk * 32 + bko);
            bw.q[0] = pb[0]; bw.q[1] = pb[1]; bw.q[2] = pb[2]; bw.q[3] = pb[3];
        } else {
#pragma unroll
            for (int j = 0; j < 16; ++j) bw.f[j] = 0.0f;
        }
        Frag b;
#pragma unroll
        for (int j = 0; j < 16; ++j) b.h[j] = (h16)bw.f[j];

#pragma unroll
        for (int mt = 0; mt < 4; ++mt) {
            const h16* ar = hlast + (size_t)(mt * 16 + m0) * HIDDEN + kk * 32 + ako;
            Frag a;
            a.q[0] = *(const u32x4*)(ar);
            a.q[1] = *(const u32x4*)(ar + 8);
            c[mt] = __builtin_amdgcn_wmma_f32_16x16x32_f16(false, a.v, false, b.v,
                                                           (short)0, c[mt], false, false);
        }
    }

    if (vok) {
        const float bias  = b2[v];
        const int   rbase = (lane < 16) ? 0 : 8;
#pragma unroll
        for (int mt = 0; mt < 4; ++mt)
#pragma unroll
            for (int r = 0; r < 8; ++r)
                out[(size_t)(mt * 16 + rbase + r) * VOCAB + v] = c[mt][r] + bias;
    }
}

// ------------------------------------------------------------------ launch --
extern "C" void kernel_launch(void* const* d_in, const int* in_sizes, int n_in,
                              void* d_out, int out_size, void* d_ws, size_t ws_size,
                              hipStream_t stream)
{
    const int*   tok = (const int*)  d_in[0];  // input      [B,S]
    const float* emb = (const float*)d_in[1];  // embed_table[V,E]
    const float* W1  = (const float*)d_in[2];  // [H, H+E]
    const float* b1  = (const float*)d_in[3];  // [H]
    const float* W2  = (const float*)d_in[4];  // [V, H]
    const float* b2  = (const float*)d_in[5];  // [V]
    float* out = (float*)d_out;

    h16* w1h   = (h16*)d_ws;                                              // 640 KB
    h16* hlast = (h16*)((char*)d_ws + (size_t)HIDDEN * KX * sizeof(h16)); // 64 KB

    const int n1 = HIDDEN * KX;
    k_w1_to_f16<<<(n1 + 255) / 256, 256, 0, stream>>>(W1, w1h, n1);
    k_rnn<<<BATCH / 16, 1024, 0, stream>>>(tok, emb, w1h, b1, hlast);
    k_logits<<<(NT2 + 7) / 8, 256, 0, stream>>>(hlast, W2, b2, out);
}
